// MaskedMultiHeadAttention_52510270160985
// MI455X (gfx1250) — matrix operand order repaired and checked by reading, not compiled, not run on hardware
//
#include <hip/hip_runtime.h>
#include <hip/hip_bf16.h>

#define BATCH 2
#define SEQ   4096
#define DIM   512
#define NH    8
#define HDIM  64

typedef __bf16 bf16;
typedef __attribute__((ext_vector_type(16))) __bf16 v16bf;
typedef __attribute__((ext_vector_type(8)))  __bf16 v8bf;
typedef __attribute__((ext_vector_type(8)))  float  v8f;

static __device__ __forceinline__ v8f wmma_bf16(v16bf a, v16bf b, v8f c) {
  // (neg_a, A, neg_b, B, c_mod, C, reuse_a, reuse_b)
  return __builtin_amdgcn_wmma_f32_16x16x32_bf16(false, a, false, b, (short)0, c, false, false);
}

// bf16 A-fragment (16x32): lanes 0-15 hold K 0-15,
// lanes 16-31 hold K 16-31; elements ascend contiguously.
static __device__ __forceinline__ v16bf load_afrag(const bf16* __restrict__ row,
                                                   int k0, int hi) {
  v8bf a = *(const v8bf*)(row + k0 + 16 * hi);
  v8bf b = *(const v8bf*)(row + k0 + 16 * hi + 8);
  v16bf f;
  #pragma unroll
  for (int i = 0; i < 8; ++i) { f[i] = a[i]; f[8 + i] = b[i]; }
  return f;
}

// ---------------------------------------------------------------------------
// Kernel 0: bulk f32 -> bf16 conversion (one pass over x and W matrices).
// ---------------------------------------------------------------------------
__global__ __launch_bounds__(256)
void cvt_bf16_kernel(const float* __restrict__ in, bf16* __restrict__ out, int n8)
{
  const int i = blockIdx.x * 256 + threadIdx.x;
  if (i >= n8) return;
  const float4* p = (const float4*)(in + (size_t)i * 8);
  const float4 a = p[0], b = p[1];
  v8bf o;
  o[0] = (bf16)a.x; o[1] = (bf16)a.y; o[2] = (bf16)a.z; o[3] = (bf16)a.w;
  o[4] = (bf16)b.x; o[5] = (bf16)b.y; o[6] = (bf16)b.z; o[7] = (bf16)b.w;
  *(v8bf*)(out + (size_t)i * 8) = o;
}

// ---------------------------------------------------------------------------
// Kernel 1: fused QKV projection, y = x @ W^T + b.  bf16 WMMA, f32 accum.
// Register-blocked: one wave computes a 32x64 output block (2 A-frags x
// 4 B-frags -> 8 WMMAs per 32-wide K step).  Q gets 1/sqrt(D) folded in;
// V is stored transposed [B,H,HD,S] for contiguous PV B-fragments.
// ---------------------------------------------------------------------------
__global__ __launch_bounds__(128)
void qkv_gemm_kernel(const bf16* __restrict__ Xb, const bf16* __restrict__ Wb,
                     const float* __restrict__ bq, const float* __restrict__ bk,
                     const float* __restrict__ bv,
                     bf16* __restrict__ Qo, bf16* __restrict__ Ko,
                     bf16* __restrict__ Vt)
{
  const int lane = threadIdx.x & 31;
  const int wid  = threadIdx.x >> 5;
  int wg = blockIdx.x * 4 + wid;
  const int nb  = wg & 7;   wg >>= 3;   // D/64 = 8  n-blocks
  const int mb  = wg & 127; wg >>= 7;   // S/32 = 128 m-blocks
  const int mat = wg % 3;               // 0=Q 1=K 2=V
  const int b   = wg / 3;

  const float* bias = (mat == 0) ? bq : (mat == 1) ? bk : bv;
  const bf16*  Wm   = Wb + (size_t)mat * DIM * DIM;

  const int lo = lane & 15;
  const int hi = lane >> 4;

  const bf16* xrow0 = Xb + ((size_t)b * SEQ + (size_t)(mb * 32 + lo)) * DIM;
  const bf16* xrow1 = xrow0 + (size_t)16 * DIM;
  const bf16* wrow[4];
  #pragma unroll
  for (int f = 0; f < 4; ++f)
    wrow[f] = Wm + (size_t)(nb * 64 + f * 16 + lo) * DIM;

  v8f acc[2][4];
  {
    const v8f zero = {};
    #pragma unroll
    for (int t = 0; t < 2; ++t)
      #pragma unroll
      for (int f = 0; f < 4; ++f) acc[t][f] = zero;
  }

  #pragma unroll 2
  for (int k0 = 0; k0 < DIM; k0 += 32) {
    const v16bf a0 = load_afrag(xrow0, k0, hi);
    const v16bf a1 = load_afrag(xrow1, k0, hi);
    #pragma unroll
    for (int f = 0; f < 4; ++f) {
      // B fragment (32x16): B[k][n] = W[n][k]; lanes 0-15 K 0-15, 16-31 K 16-31
      const v16bf bfrag = *(const v16bf*)(wrow[f] + k0 + 16 * hi);
      acc[0][f] = wmma_bf16(a0, bfrag, acc[0][f]);
      acc[1][f] = wmma_bf16(a1, bfrag, acc[1][f]);
    }
  }

  const float qscale = 0.044194173824159216f;   // 1/sqrt(512)
  #pragma unroll
  for (int f = 0; f < 4; ++f) {
    const int   n   = nb * 64 + f * 16 + lo;    // C layout: col = lane&15
    const float bia = bias[n];
    const int   h   = n >> 6;
    const int   hd  = n & 63;
    const size_t bh = (size_t)b * NH + h;
    #pragma unroll
    for (int t = 0; t < 2; ++t) {
      #pragma unroll
      for (int r = 0; r < 8; ++r) {             // C layout: row = r + 8*hi
        const int s = mb * 32 + t * 16 + r + 8 * hi;
        const float v = acc[t][f][r] + bia;
        if (mat == 0)      Qo[(bh * SEQ + s) * HDIM + hd] = (bf16)(v * qscale);
        else if (mat == 1) Ko[(bh * SEQ + s) * HDIM + hd] = (bf16)v;
        else               Vt[(bh * HDIM + hd) * SEQ + s] = (bf16)v;
      }
    }
  }
}

// ---------------------------------------------------------------------------
// Kernel 2: causal flash attention.  One wave per (b, h, 16-query tile).
// Online softmax over 32-key chunks; P re-laid out C->A via per-wave LDS.
// ---------------------------------------------------------------------------
__global__ __launch_bounds__(128)
void attn_kernel(const bf16* __restrict__ Qo, const bf16* __restrict__ Ko,
                 const bf16* __restrict__ Vt, float* __restrict__ out)
{
  __shared__ __align__(32) bf16 pl[4][16 * 32];   // 1 KB per wave

  const int lane = threadIdx.x & 31;
  const int wid  = threadIdx.x >> 5;
  int wg = blockIdx.x * 4 + wid;
  const int qt = wg & 255; wg >>= 8;    // S/16 = 256 query tiles
  const int h  = wg & 7;
  const int b  = wg >> 3;
  const int lo = lane & 15;
  const int hi = lane >> 4;

  const size_t bh = (size_t)b * NH + h;
  const bf16* Qp = Qo + bh * SEQ * HDIM;
  const bf16* Kp = Ko + bh * SEQ * HDIM;
  const bf16* Vp = Vt + bh * HDIM * SEQ;

  // Persistent Q A-fragments: 16x64 -> two 16x32 fragments.
  const bf16* qrow = Qp + (size_t)(qt * 16 + lo) * HDIM;
  const v16bf qf0 = load_afrag(qrow, 0, hi);
  const v16bf qf1 = load_afrag(qrow, 32, hi);

  float rmax[8], rsum[8];
  v8f o0 = {}, o1 = {}, o2 = {}, o3 = {};
  #pragma unroll
  for (int r = 0; r < 8; ++r) { rmax[r] = -3.0e38f; rsum[r] = 0.0f; }

  const int cmax = (qt * 16 + 15) >> 5;   // last 32-key chunk touching any row
  for (int c = 0; c <= cmax; ++c) {
    const int kb = c * 32;

    // ---- scores: two 16x16 tiles, K-dim = HD = 64 (2 WMMAs each) ----
    v8f s0 = {}, s1 = {};
    {
      v16bf kf;
      kf = *(const v16bf*)(Kp + (size_t)(kb + lo) * HDIM +       16 * hi);
      s0 = wmma_bf16(qf0, kf, s0);
      kf = *(const v16bf*)(Kp + (size_t)(kb + lo) * HDIM + 32 +  16 * hi);
      s0 = wmma_bf16(qf1, kf, s0);
      kf = *(const v16bf*)(Kp + (size_t)(kb + 16 + lo) * HDIM +       16 * hi);
      s1 = wmma_bf16(qf0, kf, s1);
      kf = *(const v16bf*)(Kp + (size_t)(kb + 16 + lo) * HDIM + 32 +  16 * hi);
      s1 = wmma_bf16(qf1, kf, s1);
    }

    if (c == cmax) {                       // causal mask only on diagonal chunk
      #pragma unroll
      for (int r = 0; r < 8; ++r) {
        const int qr = qt * 16 + r + 8 * hi;
        if (kb + lo      > qr) s0[r] = -3.0e38f;
        if (kb + 16 + lo > qr) s1[r] = -3.0e38f;
      }
    }

    // ---- online softmax per row (rows live in 16-lane halves) ----
    #pragma unroll
    for (int r = 0; r < 8; ++r) {
      float mv = fmaxf(s0[r], s1[r]);
      mv = fmaxf(mv, __shfl_xor(mv, 1, 32));
      mv = fmaxf(mv, __shfl_xor(mv, 2, 32));
      mv = fmaxf(mv, __shfl_xor(mv, 4, 32));
      mv = fmaxf(mv, __shfl_xor(mv, 8, 32));
      mv = fmaxf(mv, rmax[r]);
      const float sc = __expf(rmax[r] - mv);
      const float p0 = __expf(s0[r] - mv);
      const float p1 = __expf(s1[r] - mv);
      float ps = p0 + p1;
      ps += __shfl_xor(ps, 1, 32);
      ps += __shfl_xor(ps, 2, 32);
      ps += __shfl_xor(ps, 4, 32);
      ps += __shfl_xor(ps, 8, 32);
      rsum[r] = rsum[r] * sc + ps;
      rmax[r] = mv;
      o0[r] *= sc; o1[r] *= sc; o2[r] *= sc; o3[r] *= sc;
      const int m = r + 8 * hi;            // C-layout row
      pl[wid][m * 32 + lo]      = (bf16)p0;
      pl[wid][m * 32 + 16 + lo] = (bf16)p1;
    }

    // ---- reload P (16x32) from LDS in A-fragment layout ----
    v16bf pf;
    {
      const bf16* prow = &pl[wid][lo * 32];
      v8bf plo_ = *(const v8bf*)(prow +      16 * hi);
      v8bf phi_ = *(const v8bf*)(prow + 8 + 16 * hi);
      #pragma unroll
      for (int i = 0; i < 8; ++i) { pf[i] = plo_[i]; pf[8 + i] = phi_[i]; }
    }

    // ---- O += P @ V : B[k=key][n=hd] = Vt[hd][key] (contiguous keys) ----
    {
      const bf16* vbase = Vp + kb + 16 * hi;
      o0 = wmma_bf16(pf, *(const v16bf*)(vbase + (size_t)(lo)      * SEQ), o0);
      o1 = wmma_bf16(pf, *(const v16bf*)(vbase + (size_t)(16 + lo) * SEQ), o1);
      o2 = wmma_bf16(pf, *(const v16bf*)(vbase + (size_t)(32 + lo) * SEQ), o2);
      o3 = wmma_bf16(pf, *(const v16bf*)(vbase + (size_t)(48 + lo) * SEQ), o3);
    }

    if (c < cmax) {   // prefetch next K chunk (global_prefetch_b8)
      __builtin_prefetch(Kp + (size_t)(kb + 32 + lo) * HDIM, 0, 0);
    }
  }

  // ---- epilogue: normalize and write f32 output [B,S,D] ----
  #pragma unroll
  for (int r = 0; r < 8; ++r) {
    const float inv = 1.0f / rsum[r];
    const int s = qt * 16 + r + 8 * hi;
    float* orow = out + ((size_t)b * SEQ + s) * DIM + h * HDIM;
    orow[lo]      = o0[r] * inv;
    orow[16 + lo] = o1[r] * inv;
    orow[32 + lo] = o2[r] * inv;
    orow[48 + lo] = o3[r] * inv;
  }
}

extern "C" void kernel_launch(void* const* d_in, const int* in_sizes, int n_in,
                              void* d_out, int out_size, void* d_ws, size_t ws_size,
                              hipStream_t stream) {
  (void)in_sizes; (void)n_in; (void)out_size; (void)ws_size;
  const float* x  = (const float*)d_in[0];
  const float* Wq = (const float*)d_in[1];
  const float* bq = (const float*)d_in[2];
  const float* Wk = (const float*)d_in[3];
  const float* bk = (const float*)d_in[4];
  const float* Wv = (const float*)d_in[5];
  const float* bv = (const float*)d_in[6];
  float* out = (float*)d_out;

  // Workspace (bf16): Q [B,H,S,HD], K [B,H,S,HD], V^T [B,H,HD,S],
  //                   x  [B*S, D],  W  [3, D, D]          (~34 MB total)
  const size_t NQKV = (size_t)BATCH * SEQ * DIM;   // 4 Mi elements
  const size_t NW   = (size_t)DIM * DIM;           // 256 Ki elements
  bf16* Q  = (bf16*)d_ws;
  bf16* K  = Q  + NQKV;
  bf16* Vt = K  + NQKV;
  bf16* Xb = Vt + NQKV;
  bf16* Wb = Xb + NQKV;

  // ---- pass 0: one-time f32 -> bf16 conversion ----
  const int x8 = (int)(NQKV / 8), w8 = (int)(NW / 8);
  cvt_bf16_kernel<<<(x8 + 255) / 256, 256, 0, stream>>>(x,  Xb,          x8);
  cvt_bf16_kernel<<<(w8 + 255) / 256, 256, 0, stream>>>(Wq, Wb + 0 * NW, w8);
  cvt_bf16_kernel<<<(w8 + 255) / 256, 256, 0, stream>>>(Wk, Wb + 1 * NW, w8);
  cvt_bf16_kernel<<<(w8 + 255) / 256, 256, 0, stream>>>(Wv, Wb + 2 * NW, w8);

  // ---- pass 1: QKV projection (32x64 block per wave) ----
  const dim3 blk(128);   // 4 waves/block (wave32)
  const int waves1 = BATCH * 3 * (SEQ / 32) * (DIM / 64);   // 6144
  qkv_gemm_kernel<<<waves1 / 4, blk, 0, stream>>>(Xb, Wb, bq, bk, bv, Q, K, Vt);

  // ---- pass 2: causal flash attention ----
  const int waves2 = BATCH * NH * (SEQ / 16);               // 4096
  attn_kernel<<<waves2 / 4, blk, 0, stream>>>(Q, K, Vt, out);
}
